// MotionSampleGenerator_75273596830492
// MI455X (gfx1250) — matrix operand order repaired and checked by reading, not compiled, not run on hardware
//
#include <hip/hip_runtime.h>
#include <hip/hip_bf16.h>
#include <math.h>

// ---------------------------------------------------------------------------
// MotionSampleGenerator on gfx1250 (CDNA5, wave32, WMMA).
// GEMMs: bf16 x bf16 -> f32 accumulate via v_wmma_f32_16x16x32_bf16, with the
// block-shared A tile staged to LDS via global_load_async_to_lds_b128
// (ASYNCcnt-tracked async copy, CDNA5 §15.18.3 / CDNA5 ISA).
// Attention: flash-style online softmax, analytic ALiBi masks, WMMA for both
// q@k^T and p@v (p transposed C-layout -> A-layout through LDS).
// ---------------------------------------------------------------------------

typedef __attribute__((ext_vector_type(16))) __bf16 bfx16;
typedef __attribute__((ext_vector_type(8)))  __bf16 bfx8;
typedef __attribute__((ext_vector_type(8)))  float  f32x8;

#define NEGM (-1e9f)

__device__ __forceinline__ bfx16 cvt_bf16(const f32x8 lo, const f32x8 hi) {
  bfx16 r;
#pragma unroll
  for (int i = 0; i < 8; ++i) { r[i] = (__bf16)lo[i]; r[8 + i] = (__bf16)hi[i]; }
  return r;
}

// ---------------------------------------------------------------------------
// GEMM: out[M,N] = A[M,K] @ W[N,K]^T + bias[N]  (+ epilogue variants)
// Block = 256 threads = 8 waves; each wave owns one 16x16 C tile; all waves
// share one 16-row A tile, staged into LDS with async-to-LDS copies.
// A-operand (16x32 bf16): lane m=L&15, K pairs at 8*half and 16+8*half.
// B-operand (32x16 bf16): lane n=L&15, K at 8*half and 16+8*half, same as A.
// C layout: element e -> row e+8*half, col lane&15.
// mode: 0 plain, 1 past(2*(y)+ppe), 2 +residual, 3 relu, 4 QKV scatter
// ---------------------------------------------------------------------------
__global__ __launch_bounds__(256) void gemm_wmma_kernel(
    const float* __restrict__ A, const float* __restrict__ W,
    const float* __restrict__ bias, const float* __restrict__ resid,
    float* __restrict__ out, __bf16* __restrict__ qh, __bf16* __restrict__ kh,
    __bf16* __restrict__ vt, const float* __restrict__ pe,
    int M, int N, int K, int mode, int n_base, int Trow, int qpad, int kpad) {
  __shared__ __align__(16) float Alds[16 * 512];  // up to K=512 (32 KB)

  const int tid = threadIdx.x;
  const int lane = tid & 31;
  const int wave = tid >> 5;
  const int lm = lane & 15;
  const int half = lane >> 4;
  const int mt = blockIdx.x;
  const int nt = blockIdx.y * 8 + wave;

  // --- Stage the contiguous 16xK f32 A block into LDS (async, 16B chunks) ---
  {
    const int nchunks = K;  // K chunks * 16 B == 16*K bytes (K=256 or 512)
    const char* abase = (const char*)(A + (size_t)(mt * 16) * K);
    unsigned lds_base = (unsigned)(unsigned long long)(uintptr_t)&Alds[0];
    for (int f = tid; f < nchunks; f += 256) {
      unsigned long long gaddr = (unsigned long long)(uintptr_t)(abase + (size_t)f * 16);
      unsigned laddr = lds_base + (unsigned)f * 16u;
      asm volatile("global_load_async_to_lds_b128 %0, %1, off"
                   :: "v"(laddr), "v"(gaddr)
                   : "memory");
    }
    asm volatile("s_wait_asynccnt 0x0" ::: "memory");
    __syncthreads();
  }

  const size_t wrow = (size_t)(nt * 16 + lm) * K;

  f32x8 acc = {0.f, 0.f, 0.f, 0.f, 0.f, 0.f, 0.f, 0.f};
  for (int k0 = 0; k0 < K; k0 += 32) {
    const float* ap = &Alds[lm * K + k0];
    const float* wp = W + wrow + k0;
    __builtin_prefetch(wp + 64, 0, 1);  // global_prefetch_b8
    f32x8 alo = *(const f32x8*)(ap + 8 * half);       // ds_load
    f32x8 ahi = *(const f32x8*)(ap + 16 + 8 * half);  // ds_load
    f32x8 wlo = *(const f32x8*)(wp + 8 * half);
    f32x8 whi = *(const f32x8*)(wp + 16 + 8 * half);
    bfx16 a = cvt_bf16(alo, ahi);
    bfx16 b = cvt_bf16(wlo, whi);
    acc = __builtin_amdgcn_wmma_f32_16x16x32_bf16(false, a, false, b,
                                                  (short)0, acc, false, false);
  }

  const int n = nt * 16 + lm;
#pragma unroll
  for (int e = 0; e < 8; ++e) {
    int m = mt * 16 + e + 8 * half;
    if (m >= M) continue;
    float v = acc[e] + bias[n];
    if (mode == 0) {
      out[(size_t)m * N + n] = v;
    } else if (mode == 1) {
      out[(size_t)m * N + n] = 2.f * v + pe[(m % 25) * 256 + n];
    } else if (mode == 2) {
      out[(size_t)m * N + n] = v + resid[(size_t)m * N + n];
    } else if (mode == 3) {
      out[(size_t)m * N + n] = v > 0.f ? v : 0.f;
    } else {  // QKV scatter
      int nsem = n + n_base;
      int bb = m / Trow, t = m - bb * Trow;
      int hh = (nsem & 255) >> 5, hd = nsem & 31;
      int bh = bb * 8 + hh;
      if (nsem < 256)
        qh[((size_t)bh * qpad + t) * 32 + hd] = (__bf16)(v * 0.17677669529663689f);
      else if (nsem < 512)
        kh[((size_t)bh * kpad + t) * 32 + hd] = (__bf16)v;
      else
        vt[((size_t)bh * 32 + hd) * (size_t)kpad + t] = (__bf16)v;
    }
  }
}

// ---------------------------------------------------------------------------
// Masks (computed analytically, matching the reference):
//  mode 0: none; mode 1: tgt mask (causal + ALiBi, PERIOD=25)
//  mode 2: memory mask (lim = ratio*(i+1))
// ---------------------------------------------------------------------------
__device__ __forceinline__ float mask_score(float s, int i, int j, int Tk,
                                            int mode, int ratio, float slope) {
  if (j >= Tk) return NEGM;
  if (mode == 0) return s;
  if (mode == 1) {
    if (j > i) return NEGM;
    return s + slope * (float)(-((i - j) / 25));
  }
  int lim = ratio * (i + 1);
  if (j >= lim) return NEGM;
  return s + slope * (float)(-((lim - 1 - j) / 25));
}

// ---------------------------------------------------------------------------
// Flash attention: one wave per (b, h, 16-query tile). 32 keys / iteration.
// ---------------------------------------------------------------------------
__global__ __launch_bounds__(32) void attn_kernel(
    const __bf16* __restrict__ qh, const __bf16* __restrict__ kh,
    const __bf16* __restrict__ vt, float* __restrict__ o, int Tq, int Tk,
    int qpad, int kpad, int mask_mode, int ratio, int nqt) {
  __shared__ __align__(64) __bf16 plds[16 * 32];
  const int lane = threadIdx.x;
  const int lm = lane & 15;
  const int half = lane >> 4;
  const int qt = blockIdx.x % nqt;
  const int bh = blockIdx.x / nqt;
  const int h = bh & 7;
  const int b = bh >> 3;
  const float slope = exp2f(-(float)(h + 1));

  // q tile (A operand, 16x32), 1/sqrt(HD) already folded in.
  bfx16 aq;
  {
    const __bf16* qr = qh + ((size_t)bh * qpad + qt * 16 + lm) * 32;
    bfx8 lo = *(const bfx8*)(qr + 16 * half);
    bfx8 hi = *(const bfx8*)(qr + 16 * half + 8);
#pragma unroll
    for (int i = 0; i < 8; ++i) { aq[i] = lo[i]; aq[8 + i] = hi[i]; }
  }

  f32x8 o0 = {0.f, 0.f, 0.f, 0.f, 0.f, 0.f, 0.f, 0.f};
  f32x8 o1 = {0.f, 0.f, 0.f, 0.f, 0.f, 0.f, 0.f, 0.f};
  float mrun[8], lrun[8];
#pragma unroll
  for (int e = 0; e < 8; ++e) { mrun[e] = -1e30f; lrun[e] = 0.f; }

  const int iters = (Tk + 31) >> 5;
  for (int it = 0; it < iters; ++it) {
    const int j0 = it * 32;
    // k tiles (B operand): lane col = key j0+sub*16+lm, 16 contiguous hd.
    bfx16 bk0, bk1;
    {
      const __bf16* p0 = kh + ((size_t)bh * kpad + (j0 + lm)) * 32 + 16 * half;
      const __bf16* p1 = kh + ((size_t)bh * kpad + (j0 + 16 + lm)) * 32 + 16 * half;
      bfx8 x0 = *(const bfx8*)p0, x1 = *(const bfx8*)(p0 + 8);
      bfx8 y0 = *(const bfx8*)p1, y1 = *(const bfx8*)(p1 + 8);
#pragma unroll
      for (int i = 0; i < 8; ++i) {
        bk0[i] = x0[i]; bk0[8 + i] = x1[i];
        bk1[i] = y0[i]; bk1[8 + i] = y1[i];
      }
    }
    f32x8 z = {0.f, 0.f, 0.f, 0.f, 0.f, 0.f, 0.f, 0.f};
    f32x8 s0 = __builtin_amdgcn_wmma_f32_16x16x32_bf16(false, aq, false, bk0,
                                                       (short)0, z, false, false);
    f32x8 s1 = __builtin_amdgcn_wmma_f32_16x16x32_bf16(false, aq, false, bk1,
                                                       (short)0, z, false, false);
    // online softmax, per C-layout row e+8*half (16 lanes of a half = 16 cols)
#pragma unroll
    for (int e = 0; e < 8; ++e) {
      int i = qt * 16 + e + 8 * half;
      float sA = mask_score(s0[e], i, j0 + lm, Tk, mask_mode, ratio, slope);
      float sB = mask_score(s1[e], i, j0 + 16 + lm, Tk, mask_mode, ratio, slope);
      float mx = fmaxf(sA, sB);
#pragma unroll
      for (int d = 1; d < 16; d <<= 1) mx = fmaxf(mx, __shfl_xor(mx, d, 32));
      float mnew = fmaxf(mrun[e], mx);
      float al = __expf(mrun[e] - mnew);
      float pA = __expf(sA - mnew);
      float pB = __expf(sB - mnew);
      float rs = pA + pB;
#pragma unroll
      for (int d = 1; d < 16; d <<= 1) rs += __shfl_xor(rs, d, 32);
      lrun[e] = lrun[e] * al + rs;
      mrun[e] = mnew;
      o0[e] *= al;
      o1[e] *= al;
      plds[(e + 8 * half) * 32 + lm] = (__bf16)pA;
      plds[(e + 8 * half) * 32 + 16 + lm] = (__bf16)pB;
    }
    __syncthreads();
    // p tile back in A layout (16 queries x 32 keys)
    bfx16 ap;
    {
      const __bf16* pp = plds + lm * 32;
      bfx8 lo = *(const bfx8*)(pp + 16 * half);
      bfx8 hi = *(const bfx8*)(pp + 16 * half + 8);
#pragma unroll
      for (int i = 0; i < 8; ++i) { ap[i] = lo[i]; ap[8 + i] = hi[i]; }
    }
    // v tiles (B operand): vt is [bh, hd, kpad] so K (keys) is contiguous
    bfx16 bv0, bv1;
    {
      const __bf16* v0 = vt + ((size_t)bh * 32 + lm) * (size_t)kpad + j0 + 16 * half;
      const __bf16* v1 = vt + ((size_t)bh * 32 + 16 + lm) * (size_t)kpad + j0 + 16 * half;
      bfx8 x0 = *(const bfx8*)v0, x1 = *(const bfx8*)(v0 + 8);
      bfx8 y0 = *(const bfx8*)v1, y1 = *(const bfx8*)(v1 + 8);
#pragma unroll
      for (int i = 0; i < 8; ++i) {
        bv0[i] = x0[i]; bv0[8 + i] = x1[i];
        bv1[i] = y0[i]; bv1[8 + i] = y1[i];
      }
    }
    o0 = __builtin_amdgcn_wmma_f32_16x16x32_bf16(false, ap, false, bv0,
                                                 (short)0, o0, false, false);
    o1 = __builtin_amdgcn_wmma_f32_16x16x32_bf16(false, ap, false, bv1,
                                                 (short)0, o1, false, false);
    __syncthreads();
  }

#pragma unroll
  for (int e = 0; e < 8; ++e) {
    int t = qt * 16 + e + 8 * half;
    if (t < Tq) {
      float inv = 1.f / fmaxf(lrun[e], 1e-20f);
      size_t base = ((size_t)(b * Tq + t)) * 256 + h * 32;
      o[base + lm] = o0[e] * inv;
      o[base + 16 + lm] = o1[e] * inv;
    }
  }
}

// ---------------------------------------------------------------------------
// LayerNorm over D=256, one wave per row, in place.
// ---------------------------------------------------------------------------
__global__ __launch_bounds__(32) void ln_kernel(float* __restrict__ x,
                                                const float* __restrict__ w,
                                                const float* __restrict__ b,
                                                int M) {
  const int row = blockIdx.x;
  const int lane = threadIdx.x;
  float* xr = x + (size_t)row * 256;
  f32x8 v = *(const f32x8*)(xr + lane * 8);
  float s = 0.f;
#pragma unroll
  for (int i = 0; i < 8; ++i) s += v[i];
#pragma unroll
  for (int d = 1; d < 32; d <<= 1) s += __shfl_xor(s, d, 32);
  float mean = s * (1.f / 256.f);
  float vs = 0.f;
#pragma unroll
  for (int i = 0; i < 8; ++i) { float t = v[i] - mean; vs += t * t; }
#pragma unroll
  for (int d = 1; d < 32; d <<= 1) vs += __shfl_xor(vs, d, 32);
  float rstd = rsqrtf(vs * (1.f / 256.f) + 1e-5f);
  f32x8 wv = *(const f32x8*)(w + lane * 8);
  f32x8 bv = *(const f32x8*)(b + lane * 8);
  f32x8 r;
#pragma unroll
  for (int i = 0; i < 8; ++i) r[i] = (v[i] - mean) * rstd * wv[i] + bv[i];
  *(f32x8*)(xr + lane * 8) = r;
}

// Sinusoid table, rows 0..751 (ppe uses rows t%25; VAE uses rows 0..751).
__global__ void pe_kernel(float* __restrict__ pe) {
  int row = blockIdx.x;
  int i = threadIdx.x;  // 0..127
  float freq = __expf((float)(2 * i) * (-9.2103403719761836f / 256.0f));
  float a = (float)row * freq;
  pe[row * 256 + 2 * i] = __sinf(a);
  pe[row * 256 + 2 * i + 1] = __cosf(a);
}

__global__ void zero_kernel(uint4* __restrict__ p, int n) {
  int i = blockIdx.x * blockDim.x + threadIdx.x;
  if (i < n) p[i] = make_uint4(0u, 0u, 0u, 0u);
}

// toks = concat(mu_tok, logvar_tok, h) + sinusoid(T+2)
__global__ void toks_kernel(float* __restrict__ x, const float* __restrict__ h,
                            const float* __restrict__ mu_tok,
                            const float* __restrict__ logvar_tok,
                            const float* __restrict__ pe) {
  int idx = blockIdx.x * blockDim.x + threadIdx.x;
  if (idx >= 6016 * 256) return;
  int row = idx >> 8, d = idx & 255;
  int b = row / 752, s = row - b * 752;
  float v;
  if (s == 0) v = mu_tok[d];
  else if (s == 1) v = logvar_tok[d];
  else v = h[((size_t)(b * 750 + (s - 2))) * 256 + d];
  x[(size_t)row * 256 + d] = v + pe[s * 256 + d];
}

// motion_sample, mu, std
__global__ void final_kernel(float* __restrict__ out, const float* __restrict__ x,
                             const float* __restrict__ eps) {
  int i = blockIdx.x * blockDim.x + threadIdx.x;
  if (i >= 2048) return;
  int b = i >> 8, d = i & 255;
  float mu = x[((size_t)(b * 752)) * 256 + d];
  float lv = x[((size_t)(b * 752 + 1)) * 256 + d];
  float sd = __expf(0.5f * lv);
  out[i] = mu + sd * eps[i];
  out[2048 + i] = mu;
  out[4096 + i] = sd;
}

// ---------------------------------------------------------------------------
// Host orchestration
// ---------------------------------------------------------------------------
struct MHAp { const float *in_w, *in_b, *out_w, *out_b; };
struct DecP {
  MHAp ca, sa;
  const float *l1_w, *l1_b, *l2_w, *l2_b;
  const float *ln1w, *ln1b, *ln2w, *ln2b, *ln3w, *ln3b;
};
struct EncP {
  MHAp sa;
  const float *l1_w, *l1_b, *l2_w, *l2_b;
  const float *ln1w, *ln1b, *ln2w, *ln2b;
};

extern "C" void kernel_launch(void* const* d_in, const int* in_sizes, int n_in,
                              void* d_out, int out_size, void* d_ws, size_t ws_size,
                              hipStream_t stream) {
  (void)in_sizes; (void)n_in; (void)out_size; (void)ws_size;
  // Top-level inputs (setup_inputs dict order); params flattened as a JAX
  // pytree (keys sorted at each nesting level) after the 4 tensors.
  const float* speaker_motion = (const float*)d_in[0];  // [8,750,256]
  const float* speaker_audio  = (const float*)d_in[1];  // [8,1500,256]
  const float* past           = (const float*)d_in[2];  // [8,750,256]
  const float* eps            = (const float*)d_in[3];  // [8,256]
  auto F = [&](int i) -> const float* { return (const float*)d_in[i]; };

  auto mha_at = [&](int base) -> MHAp {  // base -> in_b, in_w, out_b, out_w
    return {F(base + 1), F(base + 0), F(base + 3), F(base + 2)};
  };
  auto dec_at = [&](int base) -> DecP {  // ca.., l1_b,l1_w,l2_b,l2_w, ln*, sa..
    DecP p;
    p.ca = mha_at(base + 0);
    p.l1_b = F(base + 4);  p.l1_w = F(base + 5);
    p.l2_b = F(base + 6);  p.l2_w = F(base + 7);
    p.ln1b = F(base + 8);  p.ln1w = F(base + 9);
    p.ln2b = F(base + 10); p.ln2w = F(base + 11);
    p.ln3b = F(base + 12); p.ln3w = F(base + 13);
    p.sa = mha_at(base + 14);
    return p;
  };
  auto enc_at = [&](int base) -> EncP {  // l1_b,l1_w,l2_b,l2_w, ln*, sa..
    EncP p;
    p.l1_b = F(base + 0); p.l1_w = F(base + 1);
    p.l2_b = F(base + 2); p.l2_w = F(base + 3);
    p.ln1b = F(base + 4); p.ln1w = F(base + 5);
    p.ln2b = F(base + 6); p.ln2w = F(base + 7);
    p.sa = mha_at(base + 8);
    return p;
  };
  DecP audio_dec  = dec_at(4);
  DecP motion_dec = dec_at(22);
  const float* past_b = F(40);
  const float* past_w = F(41);
  EncP enc0 = enc_at(42);
  EncP enc1 = enc_at(54);
  const float* logvar_tok = F(66);
  const float* mu_tok     = F(67);
  const float* proj_b     = F(68);
  const float* proj_w     = F(69);

  // Workspace layout
  char* wsb = (char*)d_ws;
  size_t off = 0;
  auto alloc = [&](size_t bytes) -> char* {
    char* p = wsb + off;
    off += (bytes + 255) & ~(size_t)255;
    return p;
  };
  float* pe   = (float*)alloc((size_t)752 * 256 * 4);
  float* x    = (float*)alloc((size_t)6016 * 256 * 4);
  float* obuf = (float*)alloc((size_t)6016 * 256 * 4);
  float* hbuf = (float*)alloc((size_t)6016 * 512 * 4);
  __bf16* qh = (__bf16*)alloc((size_t)64 * 768 * 32 * 2);
  __bf16* kh = (__bf16*)alloc((size_t)64 * 1536 * 32 * 2);
  __bf16* vt = (__bf16*)alloc((size_t)64 * 1536 * 32 * 2);

  auto gemm = [&](const float* A, const float* W, const float* bias,
                  const float* resid, float* out, int M, int N, int K, int mode,
                  int n_base, int Trow) {
    dim3 g((unsigned)(M / 16), (unsigned)(N / 128));
    gemm_wmma_kernel<<<g, 256, 0, stream>>>(A, W, bias, resid, out, qh, kh, vt,
                                            pe, M, N, K, mode, n_base, Trow,
                                            768, 1536);
  };
  auto run_attn = [&](int Tq, int Tk, int mask_mode, int ratio) {
    int nqt = (Tq + 15) / 16;
    attn_kernel<<<dim3((unsigned)(64 * nqt)), 32, 0, stream>>>(
        qh, kh, vt, obuf, Tq, Tk, 768, 1536, mask_mode, ratio, nqt);
  };
  auto run_ln = [&](const float* w, const float* b, int M) {
    ln_kernel<<<dim3((unsigned)M), 32, 0, stream>>>(x, w, b, M);
  };
  auto zero_qkv = [&]() {
    int nq = (64 * 768 * 32 * 2) / 16;
    int nk = (64 * 1536 * 32 * 2) / 16;
    zero_kernel<<<dim3((unsigned)((nq + 255) / 256)), 256, 0, stream>>>((uint4*)qh, nq);
    zero_kernel<<<dim3((unsigned)((nk + 255) / 256)), 256, 0, stream>>>((uint4*)kh, nk);
    zero_kernel<<<dim3((unsigned)((nk + 255) / 256)), 256, 0, stream>>>((uint4*)vt, nk);
  };

  // --- pipeline ---
  pe_kernel<<<752, 128, 0, stream>>>(pe);

  // x = 2*(past @ past_w^T + past_b) + ppe
  gemm(past, past_w, past_b, nullptr, x, 6000, 256, 256, 1, 0, 750);

  auto dec_layer = [&](const DecP& p, const float* mem, int Tmem, int ratio) {
    // self-attention (tmask)
    zero_qkv();
    gemm(x, p.sa.in_w, p.sa.in_b, nullptr, x, 6000, 768, 256, 4, 0, 750);
    run_attn(750, 750, 1, 0);
    gemm(obuf, p.sa.out_w, p.sa.out_b, x, x, 6000, 256, 256, 2, 0, 750);
    run_ln(p.ln1w, p.ln1b, 6000);
    // cross-attention (mmask)
    zero_qkv();
    gemm(x, p.ca.in_w, p.ca.in_b, nullptr, x, 6000, 256, 256, 4, 0, 750);
    gemm(mem, p.ca.in_w + (size_t)256 * 256, p.ca.in_b + 256, nullptr, x,
         8 * Tmem, 512, 256, 4, 256, Tmem);
    run_attn(750, Tmem, 2, ratio);
    gemm(obuf, p.ca.out_w, p.ca.out_b, x, x, 6000, 256, 256, 2, 0, 750);
    run_ln(p.ln2w, p.ln2b, 6000);
    // FFN
    gemm(x, p.l1_w, p.l1_b, nullptr, hbuf, 6000, 512, 256, 3, 0, 750);
    gemm(hbuf, p.l2_w, p.l2_b, x, x, 6000, 256, 512, 2, 0, 750);
    run_ln(p.ln3w, p.ln3b, 6000);
  };
  dec_layer(audio_dec, speaker_audio, 1500, 2);
  dec_layer(motion_dec, speaker_motion, 750, 1);

  // VAE
  gemm(x, proj_w, proj_b, nullptr, obuf, 6000, 256, 256, 0, 0, 750);
  toks_kernel<<<dim3((unsigned)((6016 * 256 + 255) / 256)), 256, 0, stream>>>(
      x, obuf, mu_tok, logvar_tok, pe);

  auto enc_layer = [&](const EncP& p) {
    zero_qkv();
    gemm(x, p.sa.in_w, p.sa.in_b, nullptr, x, 6016, 768, 256, 4, 0, 752);
    run_attn(752, 752, 0, 0);
    gemm(obuf, p.sa.out_w, p.sa.out_b, x, x, 6016, 256, 256, 2, 0, 752);
    run_ln(p.ln1w, p.ln1b, 6016);
    gemm(x, p.l1_w, p.l1_b, nullptr, hbuf, 6016, 512, 256, 3, 0, 752);
    gemm(hbuf, p.l2_w, p.l2_b, x, x, 6016, 256, 512, 2, 0, 752);
    run_ln(p.ln2w, p.ln2b, 6016);
  };
  enc_layer(enc0);
  enc_layer(enc1);

  final_kernel<<<dim3(8), 256, 0, stream>>>((float*)d_out, x, eps);
}
